// GAT_joint_5720896438727
// MI455X (gfx1250) — compile-verified, matrix operand order checked by reading, not run on hardware
//
#include <hip/hip_runtime.h>
#include <hip/hip_bf16.h>
#include <math.h>

typedef _Float16 f16;
typedef __attribute__((ext_vector_type(16))) _Float16 v16h;
typedef __attribute__((ext_vector_type(8)))  float    v8f;

#define HOUT 192
#define HEADS 3
#define HID 64
#define NEG_SLOPE 0.2f
#define EPB 4
#define MCHUNK 8   // M-tiles per wave; B fragments stay in registers across all of them

// ---------------------------------------------------------------------------
// GEMM: C[N,192] = op(A[N,K]) @ W[K,192], op = optional ReLU on A (layer fusion,
// compile-time so no per-element cndmask selects).
// Each wave owns one 16-wide N-tile and MCHUNK M-tiles. All B fragments for the
// full K extent are preloaded into registers (KSTEPS x v16h), then A tiles are
// streamed through KSTEPS unrolled v_wmma_f32_16x16x32_f16 ops.
// Fragment addressing per CDNA5 ISA 16-bit A/B layouts:
//   lane L: m/n = L&15, half = L>>4; frag elem i (0..15):
//   k = (i>>3)*16 + half*8 + (i&6) + (i&1)
// ---------------------------------------------------------------------------
template<int KSTEPS, bool RELU>
__global__ void gemm_wmma(const float* __restrict__ A, const float* __restrict__ W,
                          float* __restrict__ C, int N) {
    const int K    = KSTEPS * 32;
    const int lane = threadIdx.x & 31;
    const int wave = threadIdx.x >> 5;
    const int nTilesN = HOUT / 16;                               // 12
    const int mTiles  = (N + 15) >> 4;
    const int mChunks = (mTiles + MCHUNK - 1) / MCHUNK;
    const int task = blockIdx.x * (blockDim.x >> 5) + wave;
    if (task >= nTilesN * mChunks) return;                       // wave-uniform exit

    const int nt   = task % nTilesN;
    const int mt0  = (task / nTilesN) * MCHUNK;
    const int m    = lane & 15;
    const int half = lane >> 4;
    const int colB = nt * 16 + m;

    // ---- Preload all B fragments for this N-tile (register resident) ----
    v16h bf[KSTEPS];
#pragma unroll
    for (int s = 0; s < KSTEPS; ++s) {
#pragma unroll
        for (int i = 0; i < 16; ++i) {
            const int k = s * 32 + ((i >> 3) << 4) + (half << 3) + (i & 6) + (i & 1);
            bf[s][i] = (f16)W[(size_t)k * HOUT + colB];
        }
    }

    // ---- Stream MCHUNK A tiles through the resident B fragments ----
    for (int j = 0; j < MCHUNK; ++j) {
        const int mt = mt0 + j;
        if (mt >= mTiles) return;                                // wave-uniform
        int rowA = mt * 16 + m; if (rowA > N - 1) rowA = N - 1;  // safety clamp
        const float* Arow = A + (size_t)rowA * K;

        v8f acc = {0.f, 0.f, 0.f, 0.f, 0.f, 0.f, 0.f, 0.f};
#pragma unroll
        for (int s = 0; s < KSTEPS; ++s) {
            v16h af;
#pragma unroll
            for (int i = 0; i < 16; i += 2) {
                const int k = s * 32 + ((i >> 3) << 4) + (half << 3) + (i & 6);
                float2 fa = *(const float2*)(Arow + k);
                float ax = fa.x, ay = fa.y;
                if (RELU) { ax = fmaxf(ax, 0.f); ay = fmaxf(ay, 0.f); }
                af[i]     = (f16)ax;
                af[i + 1] = (f16)ay;
            }
            acc = __builtin_amdgcn_wmma_f32_16x16x32_f16(
                false, af, false, bf[s], (short)0, acc, false, false);
        }

        // Epilogue: uniform full-tile fast path (no per-lane EXEC churn)
        if ((mt + 1) * 16 <= N) {
            float* Crow = C + (size_t)(mt * 16 + half * 8) * HOUT + nt * 16 + m;
#pragma unroll
            for (int r = 0; r < 8; ++r)
                Crow[(size_t)r * HOUT] = acc[r];
        } else {
#pragma unroll
            for (int r = 0; r < 8; ++r) {
                const int row = mt * 16 + r + half * 8;
                if (row < N) C[(size_t)row * HOUT + nt * 16 + m] = acc[r];
            }
        }
    }
}

// ---------------------------------------------------------------------------
// Per-node attention coefficients: al_s[n,h] = <h[n,h,:], a_src[h,:]>, same dst.
// One wave per (node, head); wave32 shuffle reduction.
// ---------------------------------------------------------------------------
__global__ void attn_coef(const float* __restrict__ H, const float* __restrict__ asrc,
                          const float* __restrict__ adst, float* __restrict__ als,
                          float* __restrict__ ald, int N) {
    const int wid  = blockIdx.x * (blockDim.x >> 5) + (threadIdx.x >> 5);
    const int lane = threadIdx.x & 31;
    if (wid >= N * HEADS) return;                          // wave-uniform
    const int n = wid / HEADS, h = wid % HEADS;
    const float* hp = H + (size_t)n * HOUT + h * HID;
    const float* as = asrc + h * HID;
    const float* ad = adst + h * HID;
    float ps = hp[lane] * as[lane] + hp[lane + 32] * as[lane + 32];
    float pd = hp[lane] * ad[lane] + hp[lane + 32] * ad[lane + 32];
#pragma unroll
    for (int off = 16; off; off >>= 1) {
        ps += __shfl_xor(ps, off, 32);
        pd += __shfl_xor(pd, off, 32);
    }
    if (lane == 0) { als[n * HEADS + h] = ps; ald[n * HEADS + h] = pd; }
}

// Order-preserving f32 <-> u32 map for atomicMax on floats
__device__ __forceinline__ unsigned fenc(float x) {
    unsigned u = __float_as_uint(x);
    return (u & 0x80000000u) ? ~u : (u | 0x80000000u);
}
__device__ __forceinline__ float fdec(unsigned e) {
    unsigned u = (e & 0x80000000u) ? (e & 0x7fffffffu) : ~e;
    return __uint_as_float(u);
}

// Init per-layer state: out = bias (segment_sum + b), mx = -inf(enc=0), ssum = 0
__global__ void layer_init(float* __restrict__ outb, const float* __restrict__ bias,
                           unsigned* __restrict__ mx, float* __restrict__ ssum, int N) {
    const int i = blockIdx.x * blockDim.x + threadIdx.x;
    if (i < N * HOUT) outb[i] = bias[i % HOUT];
    if (i < N * HEADS) { mx[i] = 0u; ssum[i] = 0.f; }
}

// Pass 1: logits + segment max (atomic)
__global__ void edge_logits(const int* __restrict__ src, const int* __restrict__ dst,
                            const float* __restrict__ als, const float* __restrict__ ald,
                            float* __restrict__ elog, unsigned* __restrict__ mx,
                            int E, int ET) {
    const int e = blockIdx.x * blockDim.x + threadIdx.x;
    if (e >= ET) return;
    int s, d;
    if (e < E) { s = src[e]; d = dst[e]; } else { s = e - E; d = e - E; }
#pragma unroll
    for (int h = 0; h < HEADS; ++h) {
        float l = als[s * HEADS + h] + ald[d * HEADS + h];
        l = (l > 0.f) ? l : NEG_SLOPE * l;                 // leaky_relu
        elog[(size_t)e * HEADS + h] = l;
        atomicMax(&mx[d * HEADS + h], fenc(l));
    }
}

// Pass 2: e = exp(l - m), segment sum (atomic); elog overwritten with e
__global__ void edge_exp_sum(const int* __restrict__ dst, float* __restrict__ elog,
                             const unsigned* __restrict__ mx, float* __restrict__ ssum,
                             int E, int ET) {
    const int e = blockIdx.x * blockDim.x + threadIdx.x;
    if (e >= ET) return;
    const int d = (e < E) ? dst[e] : (e - E);
#pragma unroll
    for (int h = 0; h < HEADS; ++h) {
        const float l = elog[(size_t)e * HEADS + h];
        const float v = __expf(l - fdec(mx[d * HEADS + h]));
        elog[(size_t)e * HEADS + h] = v;
        atomicAdd(&ssum[d * HEADS + h], v);
    }
}

// Pass 3: out[dst] += alpha * h[src]  (192 threads = one feature each, EPB edges/block)
// Prefetch the next edge's source row (global_prefetch_b8) to hide gather latency
// behind the scatter atomics.
__global__ void edge_aggregate(const int* __restrict__ src, const int* __restrict__ dst,
                               const float* __restrict__ elog, const float* __restrict__ ssum,
                               const float* __restrict__ H, float* __restrict__ outb,
                               int E, int ET) {
    const int t = threadIdx.x;                             // 0..191
    const int h = t / HID;
    const int ebase = blockIdx.x * EPB;
    for (int j = 0; j < EPB; ++j) {
        const int e = ebase + j;
        if (e >= ET) return;
        int s, d;
        if (e < E) { s = src[e]; d = dst[e]; } else { s = e - E; d = e - E; }
        const int en = e + 1;
        if (en < ET) {
            const int sn = (en < E) ? src[en] : (en - E);
            __builtin_prefetch(&H[(size_t)sn * HOUT + t], 0, 0);
        }
        const float alpha = elog[(size_t)e * HEADS + h] / (ssum[d * HEADS + h] + 1e-16f);
        atomicAdd(&outb[(size_t)d * HOUT + t], H[(size_t)s * HOUT + t] * alpha);
    }
}

// Pooling
__global__ void zero_out(float* __restrict__ dout, float* __restrict__ cnt, int G) {
    const int i = blockIdx.x * blockDim.x + threadIdx.x;
    if (i < G * HOUT) dout[i] = 0.f;
    if (i < G) cnt[i] = 0.f;
}
__global__ void pool_accum(const float* __restrict__ y, const int* __restrict__ batch,
                           float* __restrict__ dout, float* __restrict__ cnt, int N) {
    const int t = threadIdx.x;                             // 192
    const int n = blockIdx.x;
    if (n >= N) return;
    const int g = batch[n];
    atomicAdd(&dout[(size_t)g * HOUT + t], fmaxf(y[(size_t)n * HOUT + t], 0.f)); // fused layer-2 ReLU
    if (t == 0) atomicAdd(&cnt[g], 1.f);
}
__global__ void pool_final(float* __restrict__ dout, const float* __restrict__ cnt, int G) {
    const int i = blockIdx.x * blockDim.x + threadIdx.x;
    if (i >= G * HOUT) return;
    const float c = fmaxf(cnt[i / HOUT], 1.f);
    dout[i] = fmaxf(dout[i] / c, 0.f);
}

extern "C" void kernel_launch(void* const* d_in, const int* in_sizes, int n_in,
                              void* d_out, int out_size, void* d_ws, size_t ws_size,
                              hipStream_t stream) {
    const float* x     = (const float*)d_in[0];
    const int*   ei    = (const int*)d_in[1];      // [2, E]
    const int*   batch = (const int*)d_in[2];      // [N]
    const float* W1    = (const float*)d_in[3];
    const float* a1s   = (const float*)d_in[4];
    const float* a1d   = (const float*)d_in[5];
    const float* b1    = (const float*)d_in[6];
    const float* W2    = (const float*)d_in[7];
    const float* a2s   = (const float*)d_in[8];
    const float* a2d   = (const float*)d_in[9];
    const float* b2    = (const float*)d_in[10];

    const int Fin = 128;
    const int N  = in_sizes[0] / Fin;              // 50000
    const int E  = in_sizes[1] / 2;                // 800000
    const int ET = E + N;                          // + self loops
    const int G  = out_size / HOUT;                // 512

    // Workspace layout (floats)
    float*    ws   = (float*)d_ws;
    float*    bufA = ws;                                   // [N,192] pre-agg features
    float*    bufB = bufA + (size_t)N * HOUT;              // [N,192] aggregated output
    float*    als  = bufB + (size_t)N * HOUT;              // [N,3]
    float*    ald  = als + (size_t)N * HEADS;              // [N,3]
    unsigned* mx   = (unsigned*)(ald + (size_t)N * HEADS); // [N,3]
    float*    ssum = (float*)(mx + (size_t)N * HEADS);     // [N,3]
    float*    elog = ssum + (size_t)N * HEADS;             // [ET,3]
    float*    cnt  = elog + (size_t)ET * HEADS;            // [G]
    float*    dout = (float*)d_out;

    const int* srcIdx = ei;
    const int* dstIdx = ei + E;

    const int mTiles  = (N + 15) / 16;
    const int mChunks = (mTiles + MCHUNK - 1) / MCHUNK;
    const int tasks   = mChunks * (HOUT / 16);
    dim3 gemmGrid((tasks + 7) / 8), gemmBlock(256);        // 8 waves/block, 1 task/wave
    const int attnBlocks = (N * HEADS + 3) / 4;
    const int initBlocks = (N * HOUT + 255) / 256;
    const int edgeBlocks = (ET + 255) / 256;
    const int aggBlocks  = (ET + EPB - 1) / EPB;

    // ---------------- Layer 1 (K = 128 -> 4 ksteps) ----------------
    gemm_wmma<4, false><<<gemmGrid, gemmBlock, 0, stream>>>(x, W1, bufA, N);
    attn_coef<<<attnBlocks, 128, 0, stream>>>(bufA, a1s, a1d, als, ald, N);
    layer_init<<<initBlocks, 256, 0, stream>>>(bufB, b1, mx, ssum, N);
    edge_logits<<<edgeBlocks, 256, 0, stream>>>(srcIdx, dstIdx, als, ald, elog, mx, E, ET);
    edge_exp_sum<<<edgeBlocks, 256, 0, stream>>>(dstIdx, elog, mx, ssum, E, ET);
    edge_aggregate<<<aggBlocks, HOUT, 0, stream>>>(srcIdx, dstIdx, elog, ssum, bufA, bufB, E, ET);

    // ---------------- Layer 2 (K = 192 -> 6 ksteps, ReLU fused into A-load) ----------------
    gemm_wmma<6, true><<<gemmGrid, gemmBlock, 0, stream>>>(bufB, W2, bufA, N);
    attn_coef<<<attnBlocks, 128, 0, stream>>>(bufA, a2s, a2d, als, ald, N);
    layer_init<<<initBlocks, 256, 0, stream>>>(bufB, b2, mx, ssum, N);
    edge_logits<<<edgeBlocks, 256, 0, stream>>>(srcIdx, dstIdx, als, ald, elog, mx, E, ET);
    edge_exp_sum<<<edgeBlocks, 256, 0, stream>>>(dstIdx, elog, mx, ssum, E, ET);
    edge_aggregate<<<aggBlocks, HOUT, 0, stream>>>(srcIdx, dstIdx, elog, ssum, bufA, bufB, E, ET);

    // ---------------- Mean pool + ReLU ----------------
    zero_out<<<(G * HOUT + 255) / 256, 256, 0, stream>>>(dout, cnt, G);
    pool_accum<<<N, HOUT, 0, stream>>>(bufB, batch, dout, cnt, N);
    pool_final<<<(G * HOUT + 255) / 256, 256, 0, stream>>>(dout, cnt, G);
}
